// USFMVisionTransformer_20126216749802
// MI455X (gfx1250) — matrix operand order repaired and checked by reading, not compiled, not run on hardware
//
#include <hip/hip_runtime.h>
#include <hip/hip_bf16.h>
#include <math.h>

// ---------------- constants ----------------
constexpr int Bb = 32, Cc = 3, IMGSZ = 224, Pp = 16, Gg = 14;
constexpr int Nn = 197;            // tokens incl cls
constexpr int NP = 224;            // padded tokens (multiple of 32)
constexpr int Dd = 768, Ll = 12, Hh = 12, HD = 64, FFd = 3072, NRd = 732;
constexpr int BH = Bb * Hh;        // 384
constexpr int Mtok = Bb * Nn;      // 6304 (multiple of 32)
constexpr int Mpatch = Bb * Gg * Gg; // 6272
constexpr float ATTN_SCALE = 0.125f; // 64^-0.5

typedef __attribute__((ext_vector_type(16))) _Float16 v16h;
typedef __attribute__((ext_vector_type(8)))  float    v8f;

union Frag16 { v16h h; unsigned int u[8]; };

// CDNA5 async global->LDS copy (ASYNCcnt-tracked, 16B per lane)
__device__ __forceinline__ void async_b128_to_lds(unsigned lds_byte_addr, const void* gaddr) {
  asm volatile("global_load_async_to_lds_b128 %0, %1, off"
               :: "v"(lds_byte_addr), "v"((unsigned long long)(size_t)gaddr)
               : "memory");
}
__device__ __forceinline__ void wait_async0() {
  asm volatile("s_wait_asynccnt 0x0" ::: "memory");
}

// ---------------- generic WMMA GEMM: C[M,Nc] = A[M,K] * Bw[Nc,K]^T + bias ----------------
// tile 32(M) x 256(N), 256 threads = 8 waves arranged 2(waveM) x 4(waveN); wave does 16x64.
__global__ __launch_bounds__(256)
void gemm_f16_wmma(const _Float16* __restrict__ A, const _Float16* __restrict__ Bw,
                   const float* __restrict__ bias, float* __restrict__ C,
                   int M, int Nc, int K) {
  __shared__ _Float16 As[32][32];    // 2 KB
  __shared__ _Float16 Bs[256][32];   // 16 KB
  const int tid = threadIdx.x;
  const int lane = tid & 31;
  const int wid = tid >> 5;
  const int waveM = wid >> 2, waveN = wid & 3;
  const int lrow = lane & 15, lhi = lane >> 4;
  const int m0 = blockIdx.y * 32;
  const int n0 = blockIdx.x * 256;

  const unsigned ldsA = (unsigned)(size_t)&As[0][0];
  const unsigned ldsB = (unsigned)(size_t)&Bs[0][0];

  v8f acc[4] = {{}, {}, {}, {}};

  // staging coordinates: A tile 32x32 halfs = 128 x b128; B tile 256x32 halfs = 1024 x b128
  const int ar = tid >> 2, ac = (tid & 3) * 8;

  for (int k0 = 0; k0 < K; k0 += 32) {
    if (tid < 128)
      async_b128_to_lds(ldsA + (unsigned)tid * 16u,
                        &A[(size_t)(m0 + ar) * K + k0 + ac]);
#pragma unroll
    for (int c = 0; c < 4; ++c) {
      const int l = tid * 8 + c * 2048;       // linear half index in B tile
      const int r = l >> 5, cc = l & 31;
      async_b128_to_lds(ldsB + (unsigned)l * 2u,
                        &Bw[(size_t)(n0 + r) * K + k0 + cc]);
    }
    wait_async0();
    __syncthreads();

    // A fragment (contiguous k order, same as B: lane row = lane%16, K halves split by lane-hi)
    Frag16 a;
    const int am = waveM * 16 + lrow;
#pragma unroll
    for (int v = 0; v < 4; ++v) {
      a.u[v]     = *(const unsigned int*)&As[am][lhi * 16 + v * 2];
      a.u[v + 4] = *(const unsigned int*)&As[am][lhi * 16 + 8 + v * 2];
    }
#pragma unroll
    for (int j = 0; j < 4; ++j) {
      Frag16 b;
      const int bn = waveN * 64 + j * 16 + lrow;
#pragma unroll
      for (int v = 0; v < 8; ++v)
        b.u[v] = *(const unsigned int*)&Bs[bn][lhi * 16 + v * 2];
      acc[j] = __builtin_amdgcn_wmma_f32_16x16x32_f16(false, a.h, false, b.h,
                                                      (short)0, acc[j], false, false);
    }
    __syncthreads();
  }

#pragma unroll
  for (int j = 0; j < 4; ++j) {
    const int col = n0 + waveN * 64 + j * 16 + lrow;
    const float bia = bias[col];
#pragma unroll
    for (int i = 0; i < 8; ++i) {
      const int row = m0 + waveM * 16 + i + 8 * lhi;
      C[(size_t)row * Nc + col] = acc[j][i] + bia;
    }
  }
}

// ---------------- attention: scores = Q*K^T + rel_pos_bias (wave per 16x16 tile) ----------------
__global__ __launch_bounds__(64)
void attn_scores_wmma(const _Float16* __restrict__ q, const _Float16* __restrict__ k,
                      const float* __restrict__ rel_table_l, const int* __restrict__ rel_index,
                      float* __restrict__ scores) {
  const int bh = blockIdx.y;
  const int h = bh % Hh;
  const int wave = threadIdx.x >> 5, lane = threadIdx.x & 31;
  const int tile = blockIdx.x * 2 + wave;       // 0..195
  const int ti = tile / (NP / 16), tj = tile % (NP / 16);
  const int lrow = lane & 15, lhi = lane >> 4;
  const _Float16* qb = q + (size_t)bh * NP * HD;
  const _Float16* kb = k + (size_t)bh * NP * HD;
  const int mi = ti * 16 + lrow;
  const int nj = tj * 16 + lrow;
  v8f acc = {};
#pragma unroll
  for (int ks = 0; ks < HD; ks += 32) {
    Frag16 a, b;
#pragma unroll
    for (int v = 0; v < 4; ++v) {
      a.u[v]     = *(const unsigned int*)&qb[(size_t)mi * HD + ks + lhi * 16 + v * 2];
      a.u[v + 4] = *(const unsigned int*)&qb[(size_t)mi * HD + ks + lhi * 16 + 8 + v * 2];
    }
#pragma unroll
    for (int v = 0; v < 8; ++v)
      b.u[v] = *(const unsigned int*)&kb[(size_t)nj * HD + ks + lhi * 16 + v * 2];
    acc = __builtin_amdgcn_wmma_f32_16x16x32_f16(false, a.h, false, b.h, (short)0, acc, false, false);
  }
  float* sb = scores + (size_t)bh * NP * NP;
#pragma unroll
  for (int i = 0; i < 8; ++i) {
    const int row = ti * 16 + i + 8 * lhi;
    float val = acc[i];
    if (row < Nn && nj < Nn)
      val += rel_table_l[(size_t)rel_index[row * Nn + nj] * Hh + h];
    sb[(size_t)row * NP + nj] = val;
  }
}

// ---------------- softmax per row; write f16 attn, zero padded cols ----------------
__global__ __launch_bounds__(256)
void softmax_rows(const float* __restrict__ scores, _Float16* __restrict__ attn) {
  const int row = blockIdx.x; // bh*NP + i
  const float* s = scores + (size_t)row * NP;
  _Float16* a = attn + (size_t)row * NP;
  __shared__ float red[256];
  const int t = threadIdx.x;
  float mx = -1e30f;
  for (int j = t; j < Nn; j += 256) mx = fmaxf(mx, s[j]);
  red[t] = mx; __syncthreads();
  for (int st = 128; st > 0; st >>= 1) { if (t < st) red[t] = fmaxf(red[t], red[t + st]); __syncthreads(); }
  mx = red[0]; __syncthreads();
  float sum = 0.f;
  for (int j = t; j < Nn; j += 256) sum += __expf(s[j] - mx);
  red[t] = sum; __syncthreads();
  for (int st = 128; st > 0; st >>= 1) { if (t < st) red[t] += red[t + st]; __syncthreads(); }
  const float inv = 1.f / red[0];
  for (int j = t; j < NP; j += 256)
    a[j] = (_Float16)((j < Nn) ? __expf(s[j] - mx) * inv : 0.f);
}

// ---------------- O = attn * V (wave per 16x16 tile) ----------------
__global__ __launch_bounds__(64)
void attn_av_wmma(const _Float16* __restrict__ attn, const _Float16* __restrict__ vmat,
                  float* __restrict__ o) {
  const int bh = blockIdx.y;
  const int wave = threadIdx.x >> 5, lane = threadIdx.x & 31;
  const int tile = blockIdx.x * 2 + wave;   // 0..55
  const int ti = tile >> 2, td = tile & 3;
  const int lrow = lane & 15, lhi = lane >> 4;
  const _Float16* ab = attn + (size_t)bh * NP * NP;
  const _Float16* vb = vmat + (size_t)bh * NP * HD;
  const int mi = ti * 16 + lrow;
  const int nd = td * 16 + lrow;
  v8f acc = {};
  for (int ks = 0; ks < NP; ks += 32) {
    Frag16 a, b;
#pragma unroll
    for (int vv = 0; vv < 4; ++vv) {
      a.u[vv]     = *(const unsigned int*)&ab[(size_t)mi * NP + ks + lhi * 16 + vv * 2];
      a.u[vv + 4] = *(const unsigned int*)&ab[(size_t)mi * NP + ks + lhi * 16 + 8 + vv * 2];
    }
#pragma unroll
    for (int vv = 0; vv < 8; ++vv) {
      const int j0 = ks + lhi * 16 + vv * 2;
      b.h[2 * vv]     = vb[(size_t)j0 * HD + nd];
      b.h[2 * vv + 1] = vb[(size_t)(j0 + 1) * HD + nd];
    }
    acc = __builtin_amdgcn_wmma_f32_16x16x32_f16(false, a.h, false, b.h, (short)0, acc, false, false);
  }
  float* ob = o + (size_t)bh * NP * HD;
#pragma unroll
  for (int i = 0; i < 8; ++i)
    ob[(size_t)(ti * 16 + i + 8 * lhi) * HD + nd] = acc[i];
}

// ---------------- layernorm over D=768, f16 output ----------------
__global__ __launch_bounds__(256)
void layernorm_f16(const float* __restrict__ x, const float* __restrict__ g,
                   const float* __restrict__ bb, _Float16* __restrict__ out) {
  const int row = blockIdx.x;
  const float* xr = x + (size_t)row * Dd;
  __shared__ float red[256];
  const int t = threadIdx.x;
  float s = 0.f;
  for (int j = t; j < Dd; j += 256) s += xr[j];
  red[t] = s; __syncthreads();
  for (int st = 128; st > 0; st >>= 1) { if (t < st) red[t] += red[t + st]; __syncthreads(); }
  const float mean = red[0] / Dd; __syncthreads();
  float s2 = 0.f;
  for (int j = t; j < Dd; j += 256) { float d = xr[j] - mean; s2 += d * d; }
  red[t] = s2; __syncthreads();
  for (int st = 128; st > 0; st >>= 1) { if (t < st) red[t] += red[t + st]; __syncthreads(); }
  const float rstd = rsqrtf(red[0] / Dd + 1e-5f);
  _Float16* orow = out + (size_t)row * Dd;
  for (int j = t; j < Dd; j += 256)
    orow[j] = (_Float16)((xr[j] - mean) * rstd * g[j] + bb[j]);
}

// ---------------- elementwise helpers ----------------
__global__ void cvt_f16_k(const float* __restrict__ in, _Float16* __restrict__ out, int n) {
  int i = blockIdx.x * 256 + threadIdx.x;
  if (i < n) out[i] = (_Float16)in[i];
}

__global__ void qkv_bias_assemble(const float* __restrict__ qb, const float* __restrict__ vb,
                                  float* __restrict__ out) {
  int i = blockIdx.x * 256 + threadIdx.x;
  if (i >= 3 * Dd) return;
  out[i] = (i < Dd) ? qb[i] : (i < 2 * Dd ? 0.f : vb[i - 2 * Dd]);
}

__global__ void qkv_repack(const float* __restrict__ qkv, _Float16* __restrict__ q,
                           _Float16* __restrict__ k, _Float16* __restrict__ v) {
  int i = blockIdx.x * 256 + threadIdx.x;
  if (i >= BH * NP * HD) return;
  const int dd = i % HD; int rest = i / HD;
  const int n = rest % NP; const int bh = rest / NP;
  const int b = bh / Hh, hh = bh % Hh;
  float qv = 0.f, kv = 0.f, vv = 0.f;
  if (n < Nn) {
    const size_t ro = (size_t)(b * Nn + n) * (3 * Dd);
    const int col = hh * HD + dd;
    qv = qkv[ro + col]; kv = qkv[ro + Dd + col]; vv = qkv[ro + 2 * Dd + col];
  }
  q[i] = (_Float16)(qv * ATTN_SCALE);
  k[i] = (_Float16)kv;
  v[i] = (_Float16)vv;
}

__global__ void o_repack(const float* __restrict__ o, _Float16* __restrict__ out) {
  int i = blockIdx.x * 256 + threadIdx.x;
  if (i >= Mtok * Dd) return;
  const int col = i % Dd, row = i / Dd;
  const int b = row / Nn, n = row % Nn;
  const int hh = col / HD, dd = col % HD;
  out[i] = (_Float16)o[((size_t)(b * Hh + hh) * NP + n) * HD + dd];
}

__global__ void residual_gamma(float* __restrict__ tok, const float* __restrict__ y,
                               const float* __restrict__ gamma) {
  int i = blockIdx.x * 256 + threadIdx.x;
  if (i >= Mtok * Dd) return;
  tok[i] += gamma[i % Dd] * y[i];
}

__global__ void gelu_f16_k(const float* __restrict__ x, _Float16* __restrict__ out, int n) {
  int i = blockIdx.x * 256 + threadIdx.x;
  if (i >= n) return;
  const float v = x[i];
  out[i] = (_Float16)(0.5f * v * (1.f + erff(v * 0.70710678118f)));
}

__global__ void im2col_f16(const float* __restrict__ x, _Float16* __restrict__ out) {
  int i = blockIdx.x * 256 + threadIdx.x;
  if (i >= Mpatch * Dd) return;
  const int col = i % Dd, row = i / Dd;
  const int b = row / (Gg * Gg), p = row % (Gg * Gg);
  const int gh = p / Gg, gw = p % Gg;
  const int c = col / (Pp * Pp), r = col % (Pp * Pp);
  const int ph = r / Pp, pw = r % Pp;
  out[i] = (_Float16)x[(((size_t)b * Cc + c) * IMGSZ + gh * Pp + ph) * IMGSZ + gw * Pp + pw];
}

__global__ void assemble_tok(const float* __restrict__ pe, const float* __restrict__ cls,
                             float* __restrict__ tok) {
  int i = blockIdx.x * 256 + threadIdx.x;
  if (i >= Mtok * Dd) return;
  const int col = i % Dd, row = i / Dd;
  const int b = row / Nn, n = row % Nn;
  tok[i] = (n == 0) ? cls[col] : pe[((size_t)b * (Gg * Gg) + (n - 1)) * Dd + col];
}

__global__ __launch_bounds__(256)
void pool_ln(const float* __restrict__ tok, const float* __restrict__ g,
             const float* __restrict__ bb, float* __restrict__ out) {
  const int b = blockIdx.x;
  __shared__ float pooled[Dd];
  __shared__ float red[256];
  const int t = threadIdx.x;
  for (int d = t; d < Dd; d += 256) {
    float s = 0.f;
    const float* base = tok + ((size_t)b * Nn + 1) * Dd + d;
    for (int n = 0; n < Gg * Gg; ++n) s += base[(size_t)n * Dd];
    pooled[d] = s / (float)(Gg * Gg);
  }
  __syncthreads();
  float s = 0.f;
  for (int d = t; d < Dd; d += 256) s += pooled[d];
  red[t] = s; __syncthreads();
  for (int st = 128; st > 0; st >>= 1) { if (t < st) red[t] += red[t + st]; __syncthreads(); }
  const float mean = red[0] / Dd; __syncthreads();
  float s2 = 0.f;
  for (int d = t; d < Dd; d += 256) { float dd = pooled[d] - mean; s2 += dd * dd; }
  red[t] = s2; __syncthreads();
  for (int st = 128; st > 0; st >>= 1) { if (t < st) red[t] += red[t + st]; __syncthreads(); }
  const float rstd = rsqrtf(red[0] / Dd + 1e-5f);
  for (int d = t; d < Dd; d += 256)
    out[b * Dd + d] = (pooled[d] - mean) * rstd * g[d] + bb[d];
}

// ---------------- host orchestration ----------------
static inline int ceil_div(long long a, long long b) { return (int)((a + b - 1) / b); }

extern "C" void kernel_launch(void* const* d_in, const int* in_sizes, int n_in,
                              void* d_out, int out_size, void* d_ws, size_t ws_size,
                              hipStream_t stream) {
  (void)in_sizes; (void)n_in; (void)out_size; (void)ws_size;
  const float* x        = (const float*)d_in[0];
  const float* patch_w  = (const float*)d_in[1];
  const float* patch_b  = (const float*)d_in[2];
  const float* cls_tok  = (const float*)d_in[3];
  const float* ln1_g    = (const float*)d_in[4];
  const float* ln1_b    = (const float*)d_in[5];
  const float* qkv_w    = (const float*)d_in[6];
  const float* q_bias   = (const float*)d_in[7];
  const float* v_bias   = (const float*)d_in[8];
  const float* rel_tab  = (const float*)d_in[9];
  const float* proj_w   = (const float*)d_in[10];
  const float* proj_b   = (const float*)d_in[11];
  const float* gamma1   = (const float*)d_in[12];
  const float* ln2_g    = (const float*)d_in[13];
  const float* ln2_b    = (const float*)d_in[14];
  const float* fc1_w    = (const float*)d_in[15];
  const float* fc1_b    = (const float*)d_in[16];
  const float* fc2_w    = (const float*)d_in[17];
  const float* fc2_b    = (const float*)d_in[18];
  const float* gamma2   = (const float*)d_in[19];
  const float* fcn_g    = (const float*)d_in[20];
  const float* fcn_b    = (const float*)d_in[21];
  const int*   rel_idx  = (const int*)d_in[22];
  float* out = (float*)d_out;

  // workspace layout (256B aligned slabs)
  char* base = (char*)d_ws;
  size_t off = 0;
  auto alloc = [&](size_t nbytes) -> char* {
    char* p = base + off;
    off += (nbytes + 255) & ~(size_t)255;
    return p;
  };
  const size_t bigN = (size_t)Mtok * FFd;                 // >= BH*NP*NP, >= Mpatch*Dd
  float*    tok   = (float*)alloc((size_t)Mtok * Dd * 4);
  _Float16* hbuf  = (_Float16*)alloc((size_t)Mtok * Dd * 2);
  float*    qkvb  = (float*)alloc((size_t)Mtok * 3 * Dd * 4);  // also reused for O fp32
  _Float16* qf    = (_Float16*)alloc((size_t)BH * NP * HD * 2);
  _Float16* kf    = (_Float16*)alloc((size_t)BH * NP * HD * 2);
  _Float16* vf    = (_Float16*)alloc((size_t)BH * NP * HD * 2);
  float*    big32 = (float*)alloc(bigN * 4);                   // scores / fc1 out / patch out
  _Float16* big16 = (_Float16*)alloc(bigN * 2);                // attn / gelu f16 / im2col A
  _Float16* wbuf  = (_Float16*)alloc((size_t)FFd * Dd * 2);    // f16 weight staging
  float*    bbuf  = (float*)alloc((size_t)3 * Dd * 4);         // qkv bias

  const dim3 blk256(256), blk64(64);

  // ---- patch embed ----
  im2col_f16<<<ceil_div((long long)Mpatch * Dd, 256), blk256, 0, stream>>>(x, big16);
  cvt_f16_k<<<ceil_div((long long)Dd * Dd, 256), blk256, 0, stream>>>(patch_w, wbuf, Dd * Dd);
  gemm_f16_wmma<<<dim3(Dd / 256, Mpatch / 32), blk256, 0, stream>>>(
      big16, wbuf, patch_b, big32, Mpatch, Dd, Dd);
  assemble_tok<<<ceil_div((long long)Mtok * Dd, 256), blk256, 0, stream>>>(big32, cls_tok, tok);

  // ---- transformer blocks ----
  for (int l = 0; l < Ll; ++l) {
    layernorm_f16<<<Mtok, blk256, 0, stream>>>(tok, ln1_g + l * Dd, ln1_b + l * Dd, hbuf);
    cvt_f16_k<<<ceil_div((long long)3 * Dd * Dd, 256), blk256, 0, stream>>>(
        qkv_w + (size_t)l * 3 * Dd * Dd, wbuf, 3 * Dd * Dd);
    qkv_bias_assemble<<<ceil_div(3 * Dd, 256), blk256, 0, stream>>>(
        q_bias + l * Dd, v_bias + l * Dd, bbuf);
    gemm_f16_wmma<<<dim3(3 * Dd / 256, Mtok / 32), blk256, 0, stream>>>(
        hbuf, wbuf, bbuf, qkvb, Mtok, 3 * Dd, Dd);
    qkv_repack<<<ceil_div((long long)BH * NP * HD, 256), blk256, 0, stream>>>(qkvb, qf, kf, vf);

    attn_scores_wmma<<<dim3((NP / 16) * (NP / 16) / 2, BH), blk64, 0, stream>>>(
        qf, kf, rel_tab + (size_t)l * NRd * Hh, rel_idx, big32);
    softmax_rows<<<BH * NP, blk256, 0, stream>>>(big32, big16);
    attn_av_wmma<<<dim3((NP / 16) * (HD / 16) / 2, BH), blk64, 0, stream>>>(big16, vf, qkvb);
    o_repack<<<ceil_div((long long)Mtok * Dd, 256), blk256, 0, stream>>>(qkvb, hbuf);

    cvt_f16_k<<<ceil_div((long long)Dd * Dd, 256), blk256, 0, stream>>>(
        proj_w + (size_t)l * Dd * Dd, wbuf, Dd * Dd);
    gemm_f16_wmma<<<dim3(Dd / 256, Mtok / 32), blk256, 0, stream>>>(
        hbuf, wbuf, proj_b + l * Dd, big32, Mtok, Dd, Dd);
    residual_gamma<<<ceil_div((long long)Mtok * Dd, 256), blk256, 0, stream>>>(
        tok, big32, gamma1 + l * Dd);

    layernorm_f16<<<Mtok, blk256, 0, stream>>>(tok, ln2_g + l * Dd, ln2_b + l * Dd, hbuf);
    cvt_f16_k<<<ceil_div((long long)FFd * Dd, 256), blk256, 0, stream>>>(
        fc1_w + (size_t)l * FFd * Dd, wbuf, FFd * Dd);
    gemm_f16_wmma<<<dim3(FFd / 256, Mtok / 32), blk256, 0, stream>>>(
        hbuf, wbuf, fc1_b + l * FFd, big32, Mtok, FFd, Dd);
    gelu_f16_k<<<ceil_div((long long)Mtok * FFd, 256), blk256, 0, stream>>>(
        big32, big16, Mtok * FFd);
    cvt_f16_k<<<ceil_div((long long)Dd * FFd, 256), blk256, 0, stream>>>(
        fc2_w + (size_t)l * Dd * FFd, wbuf, Dd * FFd);
    gemm_f16_wmma<<<dim3(Dd / 256, Mtok / 32), blk256, 0, stream>>>(
        big16, wbuf, fc2_b + l * Dd, qkvb, Mtok, Dd, FFd);
    residual_gamma<<<ceil_div((long long)Mtok * Dd, 256), blk256, 0, stream>>>(
        tok, qkvb, gamma2 + l * Dd);
  }

  // ---- mean pool over patch tokens + fc_norm ----
  pool_ln<<<Bb, blk256, 0, stream>>>(tok, fcn_g, fcn_b, out);
}
